// TemporalGRU_54425825575259
// MI455X (gfx1250) — compile-verified, matrix operand order checked by reading, not run on hardware
//
#include <hip/hip_runtime.h>
#include <hip/hip_bf16.h>
#include <math.h>

typedef __attribute__((ext_vector_type(16))) __bf16 v16bf;
typedef __attribute__((ext_vector_type(8)))  float  v8f;

#define BB 64
#define SS 512
#define DD 1024
#define HH 1024
#define LNEPS 1e-5f

// ---------------- workspace layout (bytes, 256B aligned) ----------------
static const size_t OFF_TD   = 0;                       // [B,S]   f32   131072
static const size_t OFF_XLN  = 262144;                  // [B,S,D] bf16  64 MB
static const size_t OFF_W    = OFF_XLN + (size_t)BB*SS*DD*2;  // 6 x [1024,1024] bf16 (Uz,Ur,Uh,Wz,Wr,Wh)
static const size_t OFF_HF   = OFF_W  + 6ull*HH*HH*2;   // [B,H] f32
static const size_t OFF_HBF  = OFF_HF + (size_t)BB*HH*4;// [B,H] bf16
static const size_t OFF_Z    = OFF_HBF+ (size_t)BB*HH*2;// [B,H] f32
static const size_t OFF_RH   = OFF_Z  + (size_t)BB*HH*4;// [B,H] bf16

__device__ __forceinline__ unsigned short f2bf(float f) {
  union { float f; unsigned u; } v; v.f = f;
  unsigned r = v.u + 0x7FFFu + ((v.u >> 16) & 1u);   // round-to-nearest-even
  return (unsigned short)(r >> 16);
}
__device__ __forceinline__ float sigmoidf_(float x) { return 1.0f / (1.0f + __expf(-x)); }

__device__ __forceinline__ void block_reduce2(float& a, float& b, float* sA, float* sB) {
  for (int off = 16; off > 0; off >>= 1) { a += __shfl_xor(a, off, 32); b += __shfl_xor(b, off, 32); }
  int wave = threadIdx.x >> 5, lane = threadIdx.x & 31;
  if (lane == 0) { sA[wave] = a; sB[wave] = b; }
  __syncthreads();
  if (threadIdx.x < 32) {
    float x = (lane < 8) ? sA[lane] : 0.f;
    float y = (lane < 8) ? sB[lane] : 0.f;
    for (int off = 4; off > 0; off >>= 1) { x += __shfl_xor(x, off, 32); y += __shfl_xor(y, off, 32); }
    if (lane == 0) { sA[0] = x; sB[0] = y; }
  }
  __syncthreads();
  a = sA[0]; b = sB[0];
  __syncthreads();
}

// -------- time deltas: one block per batch row --------
__global__ void __launch_bounds__(256) k_td(const float* __restrict__ ts, float* __restrict__ td) {
  __shared__ float sA[8];
  const int b = blockIdx.x;
  const float* tsr = ts + (size_t)b * SS;
  float d[2]; float mx = 0.f;
  #pragma unroll
  for (int i = 0; i < 2; ++i) {
    int s = threadIdx.x + 256 * i;
    float dd = (s == 0) ? (tsr[1] - tsr[0]) : (tsr[s] - tsr[s - 1]);
    dd = fabsf(dd) + 1e-6f;
    d[i] = dd; mx = fmaxf(mx, dd);
  }
  for (int off = 16; off > 0; off >>= 1) mx = fmaxf(mx, __shfl_xor(mx, off, 32));
  int wave = threadIdx.x >> 5, lane = threadIdx.x & 31;
  if (lane == 0) sA[wave] = mx;
  __syncthreads();
  if (threadIdx.x < 32) {
    float x = (lane < 8) ? sA[lane] : 0.f;
    for (int off = 4; off > 0; off >>= 1) x = fmaxf(x, __shfl_xor(x, off, 32));
    if (lane == 0) sA[0] = x;
  }
  __syncthreads();
  mx = sA[0];
  #pragma unroll
  for (int i = 0; i < 2; ++i) {
    int s = threadIdx.x + 256 * i;
    td[(size_t)b * SS + s] = d[i] / mx;
  }
}

// -------- input LayerNorm + f32->bf16: one block per row of [B*S, D] --------
__global__ void __launch_bounds__(256) k_ln_in(const float* __restrict__ x,
                                               const float* __restrict__ g,
                                               const float* __restrict__ bb,
                                               unsigned short* __restrict__ xln) {
  __shared__ float sA[8], sB[8];
  const size_t row = blockIdx.x;
  const float* xr = x + row * DD;
  float vals[4]; float s1 = 0.f, s2 = 0.f;
  #pragma unroll
  for (int i = 0; i < 4; ++i) {
    float v = xr[threadIdx.x + 256 * i];
    vals[i] = v; s1 += v; s2 += v * v;
  }
  block_reduce2(s1, s2, sA, sB);
  float mean = s1 * (1.0f / DD);
  float var  = s2 * (1.0f / DD) - mean * mean;
  float inv  = rsqrtf(var + LNEPS);
  #pragma unroll
  for (int i = 0; i < 4; ++i) {
    int c = threadIdx.x + 256 * i;
    xln[row * DD + c] = f2bf((vals[i] - mean) * inv * g[c] + bb[c]);
  }
}

// -------- weight f32->bf16 --------
__global__ void __launch_bounds__(256) k_cvt(const float* __restrict__ src,
                                             unsigned short* __restrict__ dst, int n) {
  for (int i = blockIdx.x * 256 + threadIdx.x; i < n; i += gridDim.x * 256)
    dst[i] = f2bf(src[i]);
}

// -------- init hidden state (f32 + bf16 copies) --------
__global__ void __launch_bounds__(256) k_initH(const float* __restrict__ h0,
                                               float* __restrict__ hf,
                                               unsigned short* __restrict__ hbf, int n) {
  int i = blockIdx.x * 256 + threadIdx.x;
  if (i < n) { float v = h0[i]; hf[i] = v; hbf[i] = f2bf(v); }
}

// -------- phase 1: z and r gates. K=2048 fused GEMM: [h | x_t] x [Uz|Wz]^T, [Ur|Wr]^T --------
// grid 32 x 256thr (8 waves). One 16x16 tile per wave; the A fragment feeds both gate WMMAs.
// K reduction split into two straight-line loops (h-part / x-part): no branches, immediate offsets.
__global__ void __launch_bounds__(256) k_phase1(const unsigned short* __restrict__ xln,
                                                const unsigned short* __restrict__ wb,
                                                const float* __restrict__ hf,
                                                const unsigned short* __restrict__ hbf,
                                                const float* __restrict__ bz,
                                                const float* __restrict__ br,
                                                const float* __restrict__ Wzt,
                                                const float* __restrict__ Wrt,
                                                const float* __restrict__ td,
                                                float* __restrict__ zout,
                                                unsigned short* __restrict__ rhout,
                                                int t) {
  const int lane  = threadIdx.x & 31;
  const int wave  = threadIdx.x >> 5;
  const int flat  = blockIdx.x * 8 + wave;
  const int mtile = flat >> 6;          // 0..3   (M = 64)
  const int ntile = flat & 63;          // 0..63  (N = 1024)
  const int arow  = mtile * 16 + (lane & 15);
  const int bcol  = ntile * 16 + (lane & 15);
  const int khalf = (lane >> 4) * 16;

  const unsigned short* Uz = wb + 0ull * HH * HH;
  const unsigned short* Ur = wb + 1ull * HH * HH;
  const unsigned short* Wz = wb + 3ull * HH * HH;
  const unsigned short* Wr = wb + 4ull * HH * HH;

  __builtin_prefetch(Uz + (size_t)bcol * HH, 0, 3);   // global_prefetch_b8
  __builtin_prefetch(Ur + (size_t)bcol * HH, 0, 3);

  const unsigned short* aph  = hbf + (size_t)arow * HH + khalf;
  const unsigned short* apx  = xln + ((size_t)arow * SS + t) * DD + khalf;
  const unsigned short* bzp  = Uz + (size_t)bcol * HH + khalf;
  const unsigned short* brp  = Ur + (size_t)bcol * HH + khalf;
  const unsigned short* bzpx = Wz + (size_t)bcol * DD + khalf;
  const unsigned short* brpx = Wr + (size_t)bcol * DD + khalf;

  v8f cz = {}; v8f cr = {};
  #pragma unroll 8
  for (int k0 = 0; k0 < HH; k0 += 32) {          // h-part
    v16bf a  = *(const v16bf*)(aph + k0);
    v16bf b0 = *(const v16bf*)(bzp + k0);
    v16bf b1 = *(const v16bf*)(brp + k0);
    cz = __builtin_amdgcn_wmma_f32_16x16x32_bf16(false, a, false, b0, (short)0, cz, false, false);
    cr = __builtin_amdgcn_wmma_f32_16x16x32_bf16(false, a, false, b1, (short)0, cr, false, false);
  }
  #pragma unroll 8
  for (int k0 = 0; k0 < DD; k0 += 32) {          // x-part
    v16bf a  = *(const v16bf*)(apx + k0);
    v16bf b0 = *(const v16bf*)(bzpx + k0);
    v16bf b1 = *(const v16bf*)(brpx + k0);
    cz = __builtin_amdgcn_wmma_f32_16x16x32_bf16(false, a, false, b0, (short)0, cz, false, false);
    cr = __builtin_amdgcn_wmma_f32_16x16x32_bf16(false, a, false, b1, (short)0, cr, false, false);
  }

  const float bzc = bz[bcol], brc = br[bcol];
  const float wzc = Wzt[bcol], wrc = Wrt[bcol];
  #pragma unroll
  for (int v = 0; v < 8; ++v) {
    int row = mtile * 16 + v + 8 * (lane >> 4);     // C/D layout: VGPR v -> M = v (+8 for hi lanes)
    float tdv = td[(size_t)row * SS + t];
    float z = sigmoidf_(cz[v] + bzc) + sigmoidf_(tdv * wzc);
    float r = sigmoidf_(cr[v] + brc) + sigmoidf_(tdv * wrc);
    zout[(size_t)row * HH + bcol]  = z;
    rhout[(size_t)row * HH + bcol] = f2bf(r * hf[(size_t)row * HH + bcol]);
  }
}

// -------- phase 2: h_tilde GEMM + gate blend + fused row LayerNorm via 64KB LDS tile --------
// grid 4 x 512thr (16 waves). Block owns 16 rows x 1024 cols. k-loop outermost, 4 accumulators
// per wave -> each A-fragment load feeds 4 WMMAs. LN: wave w normalizes tile row w.
__global__ void __launch_bounds__(512) k_phase2(const unsigned short* __restrict__ xln,
                                                const unsigned short* __restrict__ wb,
                                                const unsigned short* __restrict__ rhbf,
                                                const float* __restrict__ zbuf,
                                                const float* __restrict__ bh,
                                                const float* __restrict__ Wht,
                                                const float* __restrict__ td,
                                                const float* __restrict__ lng,
                                                const float* __restrict__ lnb,
                                                float* __restrict__ hf,
                                                unsigned short* __restrict__ hbf,
                                                float* __restrict__ outbuf,
                                                int t) {
  __shared__ float smem[16 * 1024];   // 64 KB of the WGP's 320 KB LDS
  const int lane  = threadIdx.x & 31;
  const int wave  = threadIdx.x >> 5;   // 0..15
  const int mtile = blockIdx.x;         // 0..3
  const int khalf = (lane >> 4) * 16;
  const int arow  = mtile * 16 + (lane & 15);

  const unsigned short* Uh = wb + 2ull * HH * HH;
  const unsigned short* Wh = wb + 5ull * HH * HH;
  const unsigned short* aph = rhbf + (size_t)arow * HH + khalf;
  const unsigned short* apx = xln + ((size_t)arow * SS + t) * DD + khalf;

  int bcolv[4];
  const unsigned short* bph[4];
  const unsigned short* bpx[4];
  v8f acc[4] = {};
  #pragma unroll
  for (int j = 0; j < 4; ++j) {
    int bcol = (wave * 4 + j) * 16 + (lane & 15);
    bcolv[j] = bcol;
    bph[j] = Uh + (size_t)bcol * HH + khalf;
    bpx[j] = Wh + (size_t)bcol * DD + khalf;
  }
  __builtin_prefetch(bph[0], 0, 3);

  #pragma unroll 4
  for (int k0 = 0; k0 < HH; k0 += 32) {          // rh-part
    v16bf a = *(const v16bf*)(aph + k0);
    #pragma unroll
    for (int j = 0; j < 4; ++j) {
      v16bf b = *(const v16bf*)(bph[j] + k0);
      acc[j] = __builtin_amdgcn_wmma_f32_16x16x32_bf16(false, a, false, b, (short)0, acc[j], false, false);
    }
  }
  #pragma unroll 4
  for (int k0 = 0; k0 < DD; k0 += 32) {          // x-part
    v16bf a = *(const v16bf*)(apx + k0);
    #pragma unroll
    for (int j = 0; j < 4; ++j) {
      v16bf b = *(const v16bf*)(bpx[j] + k0);
      acc[j] = __builtin_amdgcn_wmma_f32_16x16x32_bf16(false, a, false, b, (short)0, acc[j], false, false);
    }
  }

  #pragma unroll
  for (int j = 0; j < 4; ++j) {
    const int bcol = bcolv[j];
    const float bhc = bh[bcol], whc = Wht[bcol];
    #pragma unroll
    for (int v = 0; v < 8; ++v) {
      int lrow = v + 8 * (lane >> 4);
      int row  = mtile * 16 + lrow;
      float tdv  = td[(size_t)row * SS + t];
      float htil = tanhf(acc[j][v] + bhc) + tanhf(tdv * whc);
      float zv   = zbuf[(size_t)row * HH + bcol];
      float hold = hf[(size_t)row * HH + bcol];
      smem[lrow * 1024 + bcol] = (1.0f - zv) * hold + zv * htil;
    }
  }
  __syncthreads();
  // LayerNorm: wave w handles tile row w
  {
    int lrow = wave;
    int grow = mtile * 16 + lrow;
    float s1 = 0.f, s2 = 0.f;
    #pragma unroll 8
    for (int i = 0; i < 32; ++i) {
      float x = smem[lrow * 1024 + lane + 32 * i];
      s1 += x; s2 += x * x;
    }
    for (int off = 16; off > 0; off >>= 1) {
      s1 += __shfl_xor(s1, off, 32);
      s2 += __shfl_xor(s2, off, 32);
    }
    float mean = s1 * (1.0f / HH);
    float var  = s2 * (1.0f / HH) - mean * mean;
    float inv  = rsqrtf(var + LNEPS);
    #pragma unroll 8
    for (int i = 0; i < 32; ++i) {
      int c = lane + 32 * i;
      float val = (smem[lrow * 1024 + c] - mean) * inv * lng[c] + lnb[c];
      hf[(size_t)grow * HH + c]  = val;
      hbf[(size_t)grow * HH + c] = f2bf(val);
      outbuf[((size_t)grow * SS + t) * HH + c] = val;
    }
  }
}

// -------- final: outputs *= pw * decay^td, then LayerNorm (in place on d_out) --------
__global__ void __launch_bounds__(256) k_final(float* __restrict__ out,
                                               const float* __restrict__ td,
                                               const float* __restrict__ pw,
                                               const float* __restrict__ dec,
                                               const float* __restrict__ g,
                                               const float* __restrict__ bb) {
  __shared__ float sA[8], sB[8];
  const size_t row = blockIdx.x;           // row = b*S + s
  float* orow = out + row * HH;
  float scale = pw[0] * powf(dec[0], td[row]);
  float vals[4]; float s1 = 0.f, s2 = 0.f;
  #pragma unroll
  for (int i = 0; i < 4; ++i) {
    float v = orow[threadIdx.x + 256 * i] * scale;
    vals[i] = v; s1 += v; s2 += v * v;
  }
  block_reduce2(s1, s2, sA, sB);
  float mean = s1 * (1.0f / HH);
  float var  = s2 * (1.0f / HH) - mean * mean;
  float inv  = rsqrtf(var + LNEPS);
  #pragma unroll
  for (int i = 0; i < 4; ++i) {
    int c = threadIdx.x + 256 * i;
    orow[c] = (vals[i] - mean) * inv * g[c] + bb[c];
  }
}

__global__ void __launch_bounds__(256) k_copy(const float* __restrict__ src,
                                              float* __restrict__ dst, int n) {
  int i = blockIdx.x * 256 + threadIdx.x;
  if (i < n) dst[i] = src[i];
}

extern "C" void kernel_launch(void* const* d_in, const int* in_sizes, int n_in,
                              void* d_out, int out_size, void* d_ws, size_t ws_size,
                              hipStream_t stream) {
  (void)in_sizes; (void)n_in; (void)out_size; (void)ws_size;
  const float* inputs  = (const float*)d_in[0];
  const float* hidden  = (const float*)d_in[1];
  const float* tstamps = (const float*)d_in[2];
  const float* W_z = (const float*)d_in[3];  const float* b_z = (const float*)d_in[4];
  const float* U_z = (const float*)d_in[5];
  const float* W_r = (const float*)d_in[6];  const float* b_r = (const float*)d_in[7];
  const float* U_r = (const float*)d_in[8];
  const float* W_h = (const float*)d_in[9];  const float* b_h = (const float*)d_in[10];
  const float* U_h = (const float*)d_in[11];
  const float* W_zt = (const float*)d_in[12];
  const float* W_rt = (const float*)d_in[13];
  const float* W_ht = (const float*)d_in[14];
  const float* pw   = (const float*)d_in[15];
  const float* dec  = (const float*)d_in[16];
  const float* lin_g = (const float*)d_in[17]; const float* lin_b = (const float*)d_in[18];
  const float* lh_g  = (const float*)d_in[19]; const float* lh_b  = (const float*)d_in[20];
  const float* lo_g  = (const float*)d_in[21]; const float* lo_b  = (const float*)d_in[22];

  char* ws = (char*)d_ws;
  float*          td   = (float*)(ws + OFF_TD);
  unsigned short* xln  = (unsigned short*)(ws + OFF_XLN);
  unsigned short* wbf  = (unsigned short*)(ws + OFF_W);
  float*          hf   = (float*)(ws + OFF_HF);
  unsigned short* hbf  = (unsigned short*)(ws + OFF_HBF);
  float*          zbuf = (float*)(ws + OFF_Z);
  unsigned short* rhbf = (unsigned short*)(ws + OFF_RH);

  float* out = (float*)d_out;
  float* hfinal = out + (size_t)BB * SS * HH;

  // ---- setup ----
  k_td   <<<BB,      256, 0, stream>>>(tstamps, td);
  k_ln_in<<<BB * SS, 256, 0, stream>>>(inputs, lin_g, lin_b, xln);
  k_cvt  <<<1024, 256, 0, stream>>>(U_z, wbf + 0ull * HH * HH, HH * HH);
  k_cvt  <<<1024, 256, 0, stream>>>(U_r, wbf + 1ull * HH * HH, HH * HH);
  k_cvt  <<<1024, 256, 0, stream>>>(U_h, wbf + 2ull * HH * HH, HH * HH);
  k_cvt  <<<1024, 256, 0, stream>>>(W_z, wbf + 3ull * HH * HH, HH * HH);
  k_cvt  <<<1024, 256, 0, stream>>>(W_r, wbf + 4ull * HH * HH, HH * HH);
  k_cvt  <<<1024, 256, 0, stream>>>(W_h, wbf + 5ull * HH * HH, HH * HH);
  k_initH<<<(BB * HH + 255) / 256, 256, 0, stream>>>(hidden, hf, hbf, BB * HH);

  // ---- sequential scan: 2 kernels per step, all WMMA GEMMs fuse x-projection (K=2048) ----
  for (int t = 0; t < SS; ++t) {
    k_phase1<<<32, 256, 0, stream>>>(xln, wbf, hf, hbf, b_z, b_r, W_zt, W_rt, td, zbuf, rhbf, t);
    k_phase2<<<4,  512, 0, stream>>>(xln, wbf, rhbf, zbuf, b_h, W_ht, td, lh_g, lh_b, hf, hbf, out, t);
  }

  // ---- epilogue ----
  k_final<<<BB * SS, 256, 0, stream>>>(out, td, pw, dec, lo_g, lo_b);
  k_copy <<<(BB * HH + 255) / 256, 256, 0, stream>>>(hf, hfinal, BB * HH);
}
